// AligningModel_45492293599811
// MI455X (gfx1250) — matrix operand order repaired and checked by reading, not compiled, not run on hardware
//
#include <hip/hip_runtime.h>
#include <cmath>

// ---------------- types ----------------
typedef _Float16 half8 __attribute__((ext_vector_type(8)));
typedef _Float16 v16h  __attribute__((ext_vector_type(16)));
typedef float    v8f   __attribute__((ext_vector_type(8)));

// ---------------- problem constants ----------------
#define NB    32      // batch
#define TMEL  2000
#define TPH1  513     // phonemes + prepended blank
#define DM    256     // model dim
#define MELD  80
#define MEL_S 2048    // padded mel row count for half buffers
#define PH_S  640     // padded phoneme row count for half buffers (5*128)
#define SC_S  544     // padded S dim for scores/attn (513 -> 544, 17 k-steps)
#define SQRT_HALF 0.70710678118654752f

__device__ __forceinline__ v16h cat8(half8 a, half8 b) {
  return __builtin_shufflevector(a, b, 0,1,2,3,4,5,6,7,8,9,10,11,12,13,14,15);
}

#define WMMA_F16(A, B, C) \
  __builtin_amdgcn_wmma_f32_16x16x32_f16(false, (A), false, (B), (short)0, (C), false, false)

// ============================================================
// Pack conv weights [N, Din, 3] (fp32) into per-lane WMMA B-fragments (f16).
// K layout: k = t*Dk + d  (Dk = Din padded to mult of 32). Fragment order:
// frag(kstep, nblk): 32 lanes x 16 halves contiguous; lane L, elem e holds
// B[k = kstep*32 + (e/8)*16 + (L/16)*8 + (e%8)][n = nblk*16 + (L%16)]  (ISA 16-bit B layout)
// ============================================================
__global__ void pack_w_kernel(const float* __restrict__ w, _Float16* __restrict__ outp,
                              int N, int Din, int Dk) {
  int nblks = N >> 4;
  int ksteps = (3 * Dk) >> 5;
  long total = (long)ksteps * nblks * 32 * 16;
  long idx = (long)blockIdx.x * 256 + threadIdx.x;
  if (idx >= total) return;
  int e    = idx & 15;
  long r   = idx >> 4;
  int lane = r & 31;  r >>= 5;
  int nblk = (int)(r % nblks);
  int ks   = (int)(r / nblks);
  int n = nblk * 16 + (lane & 15);
  int k = ks * 32 + ((e >> 3) << 4) + ((lane >> 4) << 3) + (e & 7);
  int t = k / Dk, d = k % Dk;
  float v = (d < Din) ? w[((long)n * Din + d) * 3 + t] : 0.0f;
  outp[idx] = (_Float16)v;
}

// ============================================================
// Embedding lookup with prepended blank (id 0).
// ============================================================
__global__ void embed_kernel(const int* __restrict__ ph, const float* __restrict__ emb,
                             float* __restrict__ xph) {
  long idx = (long)blockIdx.x * 256 + threadIdx.x;
  if (idx >= (long)NB * TPH1 * DM) return;
  int d = idx & (DM - 1);
  long r = idx >> 8;
  int t = (int)(r % TPH1);
  int b = (int)(r / TPH1);
  int id = (t == 0) ? 0 : ph[(long)b * (TPH1 - 1) + (t - 1)];
  xph[idx] = emb[(long)id * DM + d];
}

// ============================================================
// Conv(k=3,'SAME') as WMMA GEMM with fully in-register GLU.
//   Template: DK (per-tap padded K), GLU, MASK (0 none, 1 t>=len, 2 t>len)
//   Block: 256 thr = 8 waves, grid (mtiles, ntiles, batch).
//   Each wave computes 16M rows and 8 accumulators:
//     GLU:  accA = "a" cols [cbase..cbase+64), accG = "g" cols (+256)
//           -> GLU + residual + sqrt(1/2) done per-lane in registers.
//     !GLU: accA = cols [wn*128 .. +64), accG = cols [wn*128+64 .. +128)
//   8 independent WMMA chains per k-step share one A fragment.
// ============================================================
template<int DK, int GLU, int MASK>
__global__ __launch_bounds__(256)
void conv_gemm_kernel(const float* __restrict__ x, int T, int Din,
                      const _Float16* __restrict__ wpack,
                      const float* __restrict__ bias,
                      const int* __restrict__ lens,
                      float* __restrict__ outp,
                      _Float16* __restrict__ out_h, int out_h_stride) {
  constexpr int KSTEPS = (3 * DK) / 32;
  constexpr int NBLKS  = (GLU ? 512 : 256) / 16;
  constexpr long BSTEP = (long)NBLKS * 512;     // halves per k-step in wpack

  extern __shared__ char smem[];
  _Float16* xs = (_Float16*)smem;         // [66][DK] staged input (halo +-1)

  const int tid = threadIdx.x;
  const int b = blockIdx.z, mtile = blockIdx.x, ntile = blockIdx.y;
  const int m0g = mtile * 64;
  const int len = (MASK && lens) ? lens[b] : 0;

  // ---- stage x tile into LDS as f16, masked & halo-padded (2 elems/iter) ----
  constexpr int HALFCNT = 66 * DK;
  for (int i = tid * 2; i < HALFCNT; i += 512) {
    int r = i / DK, c = i - r * DK;          // DK even -> c, c+1 in same row
    int gm = m0g + r - 1;
    float v0 = 0.0f, v1 = 0.0f;
    bool rowok = (gm >= 0) && (gm < T);
    if (MASK == 1 && gm >= len) rowok = false;
    if (MASK == 2 && gm >  len) rowok = false;
    if (rowok) {
      const float* xr = x + ((long)b * T + gm) * Din;
      if (c     < Din) v0 = xr[c];
      if (c + 1 < Din) v1 = xr[c + 1];
    }
    xs[i]     = (_Float16)v0;
    xs[i + 1] = (_Float16)v1;
  }
  __syncthreads();

  const int wave = tid >> 5, lane = tid & 31;
  const int wm = wave & 3, wn = wave >> 2;
  const int lm = lane & 15, hl = lane >> 4;

  // column bases (output cols) and fragment-block indices
  int cbase, nbA, nbB;
  if (GLU) { cbase = ntile * 128 + wn * 64; nbA = (cbase >> 4);       nbB = nbA + 16; }
  else     { cbase = wn * 128;              nbA = wn * 8;             nbB = nbA + 4;  }

  const _Float16* bp0 = wpack + (long)lane * 16;
  const int arow0 = wm * 16 + lm;                              // base LDS row

  auto loadA = [&](int ks) -> v16h {
    int t = (ks * 32) / DK;                     // compile-time after unroll
    int dbase = (ks * 32) - t * DK;
    const _Float16* ap = &xs[(long)(arow0 + t) * DK + dbase + hl * 8];
    return cat8(*(const half8*)ap, *(const half8*)(ap + 16));
  };

  v8f accA[4] = {}, accG[4] = {};
  v16h A = loadA(0);

#pragma unroll
  for (int ks = 0; ks < KSTEPS; ++ks) {
    const _Float16* bp = bp0 + (long)ks * BSTEP;
    if (ks + 1 < KSTEPS) __builtin_prefetch(bp + BSTEP, 0, 3);  // global_prefetch
    v16h B0[4], B1[4];
#pragma unroll
    for (int j = 0; j < 4; ++j) B0[j] = *(const v16h*)(bp + (long)(nbA + j) * 512);
#pragma unroll
    for (int j = 0; j < 4; ++j) B1[j] = *(const v16h*)(bp + (long)(nbB + j) * 512);
    v16h An = A;
    if (ks + 1 < KSTEPS) An = loadA(ks + 1);
#pragma unroll
    for (int j = 0; j < 4; ++j) accA[j] = WMMA_F16(A, B0[j], accA[j]);
#pragma unroll
    for (int j = 0; j < 4; ++j) accG[j] = WMMA_F16(A, B1[j], accG[j]);
    A = An;
  }

  if (!GLU) {
    // plain conv + bias (initial mel conv): accA cols cbase+j*16, accG +64
#pragma unroll
    for (int j = 0; j < 4; ++j)
#pragma unroll
      for (int r = 0; r < 8; ++r) {
        int gm = m0g + wm * 16 + r + hl * 8;
        if (gm < T) {
          int c0 = cbase + j * 16 + lm;
          outp[((long)b * T + gm) * DM + c0]      = accA[j][r] + bias[c0];
          outp[((long)b * T + gm) * DM + c0 + 64] = accG[j][r] + bias[c0 + 64];
        }
      }
  } else {
    // in-register GLU: out = (a*sigmoid(g) + x) * sqrt(0.5)
#pragma unroll
    for (int j = 0; j < 4; ++j)
#pragma unroll
      for (int r = 0; r < 8; ++r) {
        int lrow = wm * 16 + r + hl * 8;          // local row in [0,64)
        int gm = m0g + lrow;
        if (gm < T) {
          int gc = cbase + j * 16 + lm;
          float a = accA[j][r] + bias[gc];
          float g = accG[j][r] + bias[256 + gc];
          float xr = (float)xs[(long)(lrow + 1) * DK + gc];   // already masked
          float o = (a * (1.0f / (1.0f + expf(-g))) + xr) * SQRT_HALF;
          outp[((long)b * T + gm) * DM + gc] = o;
          if (out_h) out_h[((long)b * out_h_stride + gm) * DM + gc] = (_Float16)o;
        }
      }
  }
}

// ============================================================
// ph_sq[b,s] = sum_d ph_enc[b,s,d]^2   (1 wave per row)
// ============================================================
__global__ __launch_bounds__(256)
void rowsq_kernel(const float* __restrict__ x, float* __restrict__ sq) {
  int row = blockIdx.x * 8 + (threadIdx.x >> 5);
  int lane = threadIdx.x & 31;
  if (row >= NB * TPH1) return;
  int b = row / TPH1, s = row - b * TPH1;
  const float* p = x + (long)row * DM;
  float acc = 0.0f;
  for (int d = lane; d < DM; d += 32) { float v = p[d]; acc += v * v; }
#pragma unroll
  for (int m = 16; m; m >>= 1) acc += __shfl_xor(acc, m, 32);
  if (lane == 0) sq[(long)b * SC_S + s] = acc;
}

// ============================================================
// scores[b,t,s] = 2*<mel_enc[b,t], ph_enc[b,s]> - ph_sq[b,s], masked s>len.
// (mel_sq is constant per row -> cancels under softmax; omitted.)
// WMMA GEMM M=2048 N=640 K=256. Wave tile 32M x 64N: two A fragments
// share four B fragments -> 8 WMMAs per K-step per wave.
// ============================================================
__global__ __launch_bounds__(256)
void dist_gemm_kernel(const _Float16* __restrict__ melh, const _Float16* __restrict__ phh,
                      const float* __restrict__ phsq, const int* __restrict__ plens,
                      float* __restrict__ scores) {
  const int b = blockIdx.z, mtile = blockIdx.x, ntile = blockIdx.y;
  const int wave = threadIdx.x >> 5, lane = threadIdx.x & 31;
  const int wm = wave & 3, wn = wave >> 2;
  const int lm = lane & 15, hl = lane >> 4;
  const int m0 = mtile * 128 + wm * 32;
  const int n0 = ntile * 128 + wn * 64;
  const int len = plens[b];

  const _Float16* arow0 = melh + ((long)b * MEL_S + m0 + lm) * DM + hl * 16;
  const _Float16* arow1 = arow0 + 16 * DM;
  const _Float16* brow[4];
#pragma unroll
  for (int j = 0; j < 4; ++j)
    brow[j] = phh + ((long)b * PH_S + n0 + j * 16 + lm) * DM + hl * 16;

  v8f acc0[4] = {}, acc1[4] = {};
#pragma unroll
  for (int ks = 0; ks < 8; ++ks) {
    int kk = ks * 32;
    v16h A0 = cat8(*(const half8*)(arow0 + kk), *(const half8*)(arow0 + kk + 8));
    v16h A1 = cat8(*(const half8*)(arow1 + kk), *(const half8*)(arow1 + kk + 8));
    v16h Bf[4];
#pragma unroll
    for (int j = 0; j < 4; ++j) Bf[j] = *(const v16h*)(brow[j] + kk);
#pragma unroll
    for (int j = 0; j < 4; ++j) acc0[j] = WMMA_F16(A0, Bf[j], acc0[j]);
#pragma unroll
    for (int j = 0; j < 4; ++j) acc1[j] = WMMA_F16(A1, Bf[j], acc1[j]);
  }

#pragma unroll
  for (int g = 0; g < 2; ++g)
#pragma unroll
    for (int j = 0; j < 4; ++j)
#pragma unroll
      for (int r = 0; r < 8; ++r) {
        int gm = m0 + g * 16 + r + hl * 8;
        int s = n0 + j * 16 + lm;
        float av = g ? acc1[j][r] : acc0[j][r];
        if (gm < TMEL && s < TPH1) {
          float v = 2.0f * av - phsq[(long)b * SC_S + s];
          if (s > len) v = -1e9f;
          scores[((long)b * TMEL + gm) * SC_S + s] = v;
        }
      }
}

// ============================================================
// Row softmax over s<513, write f16 attn (K padded to 544 with zeros).
// 1 wave per row.
// ============================================================
__global__ __launch_bounds__(256)
void softmax_kernel(const float* __restrict__ scores, _Float16* __restrict__ attn) {
  int row = blockIdx.x * 8 + (threadIdx.x >> 5);
  int lane = threadIdx.x & 31;
  if (row >= NB * TMEL) return;
  int b = row / TMEL, t = row - b * TMEL;
  const float* sp = scores + (long)row * SC_S;
  _Float16* arow = attn + ((long)b * MEL_S + t) * SC_S;

  float v[17];
  float mx = -3.0e38f;
#pragma unroll
  for (int i = 0; i < 17; ++i) {
    int s = lane + i * 32;
    float z = (s < TPH1) ? sp[s] : -3.0e38f;
    v[i] = z; if (z > mx) mx = z;
  }
#pragma unroll
  for (int m = 16; m; m >>= 1) { float o = __shfl_xor(mx, m, 32); if (o > mx) mx = o; }
  float sum = 0.0f;
#pragma unroll
  for (int i = 0; i < 17; ++i) { float e = expf(v[i] - mx); v[i] = e; sum += e; }
#pragma unroll
  for (int m = 16; m; m >>= 1) sum += __shfl_xor(sum, m, 32);
  float inv = 1.0f / sum;
#pragma unroll
  for (int i = 0; i < 17; ++i) {
    int s = lane + i * 32;
    if (s < TPH1) arow[s] = (_Float16)(v[i] * inv);
  }
  if (lane < SC_S - TPH1) arow[TPH1 + lane] = (_Float16)0.0f;   // zero K-pad
}

// ============================================================
// ph_encT[b,d,s] = ph_enc_h[b,s,d] (s<513), zero-padded to 544.
// ============================================================
__global__ void transpose_ph_kernel(const _Float16* __restrict__ phh, _Float16* __restrict__ phT) {
  long idx = (long)blockIdx.x * 256 + threadIdx.x;
  if (idx >= (long)NB * DM * SC_S) return;
  int s = (int)(idx % SC_S);
  long r = idx / SC_S;
  int d = (int)(r & (DM - 1));
  int b = (int)(r >> 8);
  phT[idx] = (s < TPH1) ? phh[((long)b * PH_S + s) * DM + d] : (_Float16)0.0f;
}

// ============================================================
// context = attn @ ph_enc : M=2048 N=256 K=544; writes out[...,256:512].
// Wave tile 32M x 64N like dist_gemm.
// ============================================================
__global__ __launch_bounds__(256)
void ctx_gemm_kernel(const _Float16* __restrict__ attn, const _Float16* __restrict__ phT,
                     float* __restrict__ outp) {
  const int b = blockIdx.z, mtile = blockIdx.x, ntile = blockIdx.y;
  const int wave = threadIdx.x >> 5, lane = threadIdx.x & 31;
  const int wm = wave & 3, wn = wave >> 2;
  const int lm = lane & 15, hl = lane >> 4;
  const int m0 = mtile * 128 + wm * 32;
  const int n0 = ntile * 128 + wn * 64;

  const _Float16* arow0 = attn + ((long)b * MEL_S + m0 + lm) * SC_S + hl * 16;
  const _Float16* arow1 = arow0 + 16 * SC_S;
  const _Float16* brow[4];
#pragma unroll
  for (int j = 0; j < 4; ++j)
    brow[j] = phT + ((long)b * DM + n0 + j * 16 + lm) * SC_S + hl * 16;

  v8f acc0[4] = {}, acc1[4] = {};
#pragma unroll
  for (int ks = 0; ks < 17; ++ks) {
    int kk = ks * 32;
    v16h A0 = cat8(*(const half8*)(arow0 + kk), *(const half8*)(arow0 + kk + 8));
    v16h A1 = cat8(*(const half8*)(arow1 + kk), *(const half8*)(arow1 + kk + 8));
    v16h Bf[4];
#pragma unroll
    for (int j = 0; j < 4; ++j) Bf[j] = *(const v16h*)(brow[j] + kk);
#pragma unroll
    for (int j = 0; j < 4; ++j) acc0[j] = WMMA_F16(A0, Bf[j], acc0[j]);
#pragma unroll
    for (int j = 0; j < 4; ++j) acc1[j] = WMMA_F16(A1, Bf[j], acc1[j]);
  }

#pragma unroll
  for (int g = 0; g < 2; ++g)
#pragma unroll
    for (int j = 0; j < 4; ++j)
#pragma unroll
      for (int r = 0; r < 8; ++r) {
        int gm = m0 + g * 16 + r + hl * 8;
        int c = n0 + j * 16 + lm;
        float av = g ? acc1[j][r] : acc0[j][r];
        if (gm < TMEL) outp[((long)b * TMEL + gm) * 512 + DM + c] = av;
      }
}

// ============================================================
// out[...,0:256] = mel_enc (fp32)
// ============================================================
__global__ void copy_mel_kernel(const float* __restrict__ mel, float* __restrict__ outp) {
  long idx = (long)blockIdx.x * 256 + threadIdx.x;
  if (idx >= (long)NB * TMEL * DM) return;
  int d = (int)(idx & (DM - 1));
  long row = idx >> 8;
  outp[row * 512 + d] = mel[idx];
}

// ============================================================
// host launcher
// ============================================================
extern "C" void kernel_launch(void* const* d_in, const int* in_sizes, int n_in,
                              void* d_out, int out_size, void* d_ws, size_t ws_size,
                              hipStream_t stream) {
  (void)in_sizes; (void)n_in; (void)out_size; (void)ws_size;
  const float* mels       = (const float*)d_in[0];
  const int*   phonemes   = (const int*)  d_in[1];
  const int*   mel_lens   = (const int*)  d_in[2];
  const int*   ph_lens    = (const int*)  d_in[3];
  const float* embedding  = (const float*)d_in[4];
  const float* mel_conv_w = (const float*)d_in[5];
  const float* mel_conv_b = (const float*)d_in[6];
  const float* ph_w       = (const float*)d_in[7];
  const float* ph_b       = (const float*)d_in[8];
  const float* mel_w      = (const float*)d_in[9];
  const float* mel_b      = (const float*)d_in[10];
  float* outp = (float*)d_out;

  char* ws = (char*)d_ws; size_t cur = 0;
  auto alloc = [&](size_t bytes) -> void* {
    void* p = ws + cur; cur = (cur + bytes + 255) & ~(size_t)255; return p;
  };

  float* melA = (float*)alloc((size_t)NB * TMEL * DM * 4);
  float* melB = (float*)alloc((size_t)NB * TMEL * DM * 4);
  float* phA  = (float*)alloc((size_t)NB * TPH1 * DM * 4);
  float* phB  = (float*)alloc((size_t)NB * TPH1 * DM * 4);
  _Float16* wconv = (_Float16*)alloc((size_t)9 * 16 * 32 * 16 * 2);
  _Float16* wmel[4]; _Float16* wph[4];
  for (int i = 0; i < 4; ++i) wmel[i] = (_Float16*)alloc((size_t)24 * 32 * 32 * 16 * 2);
  for (int i = 0; i < 4; ++i) wph[i]  = (_Float16*)alloc((size_t)24 * 32 * 32 * 16 * 2);
  _Float16* melh = (_Float16*)alloc((size_t)NB * MEL_S * DM * 2);
  _Float16* phh  = (_Float16*)alloc((size_t)NB * PH_S * DM * 2);
  _Float16* phT  = (_Float16*)alloc((size_t)NB * DM * SC_S * 2);
  float* scores  = (float*)alloc((size_t)NB * TMEL * SC_S * 4);
  _Float16* attn = (_Float16*)alloc((size_t)NB * MEL_S * SC_S * 2);
  float* phsq    = (float*)alloc((size_t)NB * SC_S * 4);

  // ---- pack weights to f16 WMMA fragments ----
  {
    long tot = (long)9 * 16 * 32 * 16;
    pack_w_kernel<<<(int)((tot + 255) / 256), 256, 0, stream>>>(mel_conv_w, wconv, 256, MELD, 96);
    tot = (long)24 * 32 * 32 * 16;
    for (int i = 0; i < 4; ++i)
      pack_w_kernel<<<(int)((tot + 255) / 256), 256, 0, stream>>>(
          mel_w + (size_t)i * 512 * 256 * 3, wmel[i], 512, DM, DM);
    for (int i = 0; i < 4; ++i)
      pack_w_kernel<<<(int)((tot + 255) / 256), 256, 0, stream>>>(
          ph_w + (size_t)i * 512 * 256 * 3, wph[i], 512, DM, DM);
  }

  // ---- phoneme embedding ----
  {
    long tot = (long)NB * TPH1 * DM;
    embed_kernel<<<(int)((tot + 255) / 256), 256, 0, stream>>>(phonemes, embedding, phA);
  }

  // ---- initial mel conv (80 -> 256, no mask, no GLU): block covers all 256 cols ----
  {
    size_t sh = (size_t)66 * 96 * 2;
    conv_gemm_kernel<96, 0, 0><<<dim3(32, 1, NB), 256, sh, stream>>>(
        mels, TMEL, MELD, wconv, mel_conv_b, nullptr, melA, nullptr, 0);
  }

  // ---- 4 mel GLU blocks (ping-pong melA <-> melB) ----
  {
    size_t sh = (size_t)66 * DM * 2;
    float* bufs[2] = { melA, melB };
    for (int i = 0; i < 4; ++i) {
      float* xin = bufs[i & 1];
      float* xout = bufs[(i + 1) & 1];
      conv_gemm_kernel<DM, 1, 1><<<dim3(32, 2, NB), 256, sh, stream>>>(
          xin, TMEL, DM, wmel[i], mel_b + (size_t)i * 512,
          mel_lens, xout, (i == 3) ? melh : nullptr, MEL_S);
    }
  }

  // ---- 4 phoneme GLU blocks (ping-pong phA <-> phB) ----
  {
    size_t sh = (size_t)66 * DM * 2;
    float* bufs[2] = { phA, phB };
    for (int i = 0; i < 4; ++i) {
      float* xin = bufs[i & 1];
      float* xout = bufs[(i + 1) & 1];
      conv_gemm_kernel<DM, 1, 2><<<dim3(9, 2, NB), 256, sh, stream>>>(
          xin, TPH1, DM, wph[i], ph_b + (size_t)i * 512,
          ph_lens, xout, (i == 3) ? phh : nullptr, PH_S);
    }
  }
  // final fp32 encoders: mel in melA, phoneme in phA

  // ---- ph_sq ----
  rowsq_kernel<<<(NB * TPH1 + 7) / 8, 256, 0, stream>>>(phA, phsq);

  // ---- distances (masked) ----
  dist_gemm_kernel<<<dim3(16, 5, NB), 256, 0, stream>>>(melh, phh, phsq, ph_lens, scores);

  // ---- softmax -> f16 attn ----
  softmax_kernel<<<(NB * TMEL + 7) / 8, 256, 0, stream>>>(scores, attn);

  // ---- transpose ph_enc for context GEMM B-fragments ----
  {
    long tot = (long)NB * DM * SC_S;
    transpose_ph_kernel<<<(int)((tot + 255) / 256), 256, 0, stream>>>(phh, phT);
  }

  // ---- context GEMM -> out[...,256:512] ----
  ctx_gemm_kernel<<<dim3(16, 2, NB), 256, 0, stream>>>(attn, phT, outp);

  // ---- out[...,0:256] = mel_enc ----
  {
    long tot = (long)NB * TMEL * DM;
    copy_mel_kernel<<<(int)((tot + 255) / 256), 256, 0, stream>>>(melA, outp);
  }
}
